// DGCNN_28071906246657
// MI455X (gfx1250) — matrix operand order repaired and checked by reading, not compiled, not run on hardware
//
#include <hip/hip_runtime.h>

#define NB    8
#define NPTS  2048
#define KNN   20
#define KNN1  21
#define CATC  512
#define GT    2            // row tiles per GEMM block (share one B tile)
#define NEG_SLOPE 0.2f
#define BN_EPS    1e-5f
#define NEG_INF  (-3.402823466e38f)

typedef __attribute__((ext_vector_type(16))) __bf16          v16bf;
typedef __attribute__((ext_vector_type(16))) unsigned short  v16u;
typedef __attribute__((ext_vector_type(8)))  float           v8f;

// float -> bf16 bits, round-to-nearest-even
__device__ inline unsigned short f2bf(float f) {
  unsigned u = __float_as_uint(f);
  unsigned r = u + 0x7FFFu + ((u >> 16) & 1u);
  return (unsigned short)(r >> 16);
}

// Monotonic float atomic max (targets initialized with NEG_INF)
__device__ inline void atomicMaxF(float* addr, float val) {
  if (val >= 0.f) atomicMax((int*)addr, __float_as_int(val));
  else            atomicMin((unsigned int*)addr, __float_as_uint(val));
}

// ---------------------------------------------------------------------------
// One 16x16x32 bf16 WMMA step.
// A staged in LDS row-major [16][Apitch] (bf16 bits), Bp row-major [32][16].
// Per ISA 7.12.2: A 16-bit 16x32: lanes 0-15 -> M=lane, K in {0..7,16..23};
// lanes 16-31 -> M=lane-16, K in {8..15,24..31}. B 32x16: N=lane%16, K per element as for A.
// ---------------------------------------------------------------------------
__device__ inline v8f wmma_tile(const unsigned short* Ap, int Apitch,
                                const unsigned short* Bp, v8f acc, int lane) {
  const int row   = lane & 15;
  const int kbase = (lane < 16) ? 0 : 8;
  v16u au, bu;
  const unsigned short* ar = Ap + row * Apitch;
#pragma unroll
  for (int j = 0; j < 8; ++j) { au[j] = ar[kbase + j]; au[j + 8] = ar[16 + kbase + j]; }
  const unsigned short* br = Bp + row;
#pragma unroll
  for (int j = 0; j < 8; ++j) { bu[j] = br[(kbase + j) * 16]; bu[j + 8] = br[(16 + kbase + j) * 16]; }
  v16bf a = __builtin_bit_cast(v16bf, au);
  v16bf b = __builtin_bit_cast(v16bf, bu);
  return __builtin_amdgcn_wmma_f32_16x16x32_bf16(false, a, false, b, (short)0, acc,
                                                 false, false);
}

// Async-copy one B tile (Ktot rows x 16 cols bf16, row = 32 contiguous bytes)
// from transposed weights Wt[k][O] into LDS, via gfx1250 async-to-LDS DMA.
// 2 lanes per row x 16 rows per instruction; tracked by ASYNCcnt.
__device__ inline void async_copy_B(const unsigned short* Wt, int O, int ot,
                                    unsigned ldsB, int Ktot, int lane) {
  // prior wave DS reads of this LDS region are unordered vs async writes
  asm volatile("s_wait_dscnt 0x0" ::: "memory");
  for (int krb = 0; krb < Ktot; krb += 16) {
    int kr = krb + (lane >> 1);
    if (kr < Ktot) {
      unsigned goff = (unsigned)((kr * O + ot * 16) * 2 + (lane & 1) * 16);
      unsigned loff = ldsB + (unsigned)(kr * 32 + (lane & 1) * 16);
      asm volatile("global_load_async_to_lds_b128 %0, %1, %2"
                   :: "v"(loff), "v"(goff), "s"(Wt) : "memory");
    }
  }
  asm volatile("s_wait_asynccnt 0x0" ::: "memory");
}

// ---------------------------------------------------------------------------
// x (B,N,3) -> x0 (B,3,N) channel-major
// ---------------------------------------------------------------------------
__global__ void xpose_kernel(const float* __restrict__ x, float* __restrict__ x0) {
  int t = blockIdx.x * blockDim.x + threadIdx.x;
  if (t >= NB * NPTS) return;
  int b = t / NPTS, n = t - b * NPTS;
#pragma unroll
  for (int c = 0; c < 3; ++c)
    x0[((size_t)b * 3 + c) * NPTS + n] = x[((size_t)b * NPTS + n) * 3 + c];
}

__global__ void fill_kernel(float* __restrict__ p, float v, int n) {
  int i = blockIdx.x * blockDim.x + threadIdx.x;
  if (i < n) p[i] = v;
}

// W (O x Ktot) f32 -> Wt (Ktot x O) bf16 (transposed for contiguous B rows)
__global__ void cvtw_t_kernel(const float* __restrict__ W, unsigned short* __restrict__ Wt,
                              int O, int Ktot) {
  int i = blockIdx.x * blockDim.x + threadIdx.x;
  if (i >= O * Ktot) return;
  int o = i / Ktot, c = i - o * Ktot;
  Wt[c * O + o] = f2bf(W[i]);
}

// ---------------------------------------------------------------------------
// kNN: one wave per (b,n). dist = 2*x_n.x_m - |x_n|^2 - |x_m|^2, top-21,
// drop first (self), store 20 indices. Per-lane sorted lists + wave merge.
// ---------------------------------------------------------------------------
__global__ __launch_bounds__(32)
void knn_kernel(const float* __restrict__ X, int cin, int coffin, int bstride,
                int* __restrict__ idxout) {
  const int b    = blockIdx.x / NPTS;
  const int n    = blockIdx.x - b * NPTS;
  const int lane = threadIdx.x;
  __shared__ float ctr[256];
  __shared__ float ld[32][KNN1];
  __shared__ int   li[32][KNN1];
  const float* Xb = X + (size_t)b * bstride + (size_t)coffin * NPTS;

  for (int c = lane; c < cin; c += 32) ctr[c] = Xb[c * NPTS + n];
  __syncthreads();
  float sqn = 0.f;
  for (int c = 0; c < cin; ++c) sqn += ctr[c] * ctr[c];

  for (int j = 0; j < KNN1; ++j) { ld[lane][j] = NEG_INF; li[lane][j] = 0; }
  for (int m = lane; m < NPTS; m += 32) {
    if (m + 32 < NPTS) __builtin_prefetch(Xb + m + 32, 0, 0);  // global_prefetch
    float dot = 0.f, sqm = 0.f;
    for (int c = 0; c < cin; ++c) {
      float xm = Xb[c * NPTS + m];
      dot += xm * ctr[c];
      sqm += xm * xm;
    }
    float dist = 2.f * dot - sqn - sqm;
    if (dist > ld[lane][KNN1 - 1]) {
      int p = KNN1 - 1;
      while (p > 0 && ld[lane][p - 1] < dist) {
        ld[lane][p] = ld[lane][p - 1]; li[lane][p] = li[lane][p - 1]; --p;
      }
      ld[lane][p] = dist; li[lane][p] = m;
    }
  }
  __syncthreads();

  int p = 0;
  for (int j = 0; j < KNN1; ++j) {
    float v = (p < KNN1) ? ld[lane][p] : NEG_INF;
    int   h = (p < KNN1) ? li[lane][p] : 0;
    float mx = v;
#pragma unroll
    for (int off = 16; off > 0; off >>= 1) mx = fmaxf(mx, __shfl_xor(mx, off, 32));
    unsigned long long bal = __ballot(v == mx);
    int src  = __ffsll(bal) - 1;
    int widx = __shfl(h, src, 32);
    if (lane == src) ++p;
    if (j > 0 && lane == 0) idxout[((size_t)b * NPTS + n) * KNN + (j - 1)] = widx;
  }
}

// ---------------------------------------------------------------------------
// Edge-conv GEMM: one wave per GT row tiles (rows = (b,n,k) edges).
// B tile async-DMA'd to LDS once per 16-channel group, shared by GT tiles.
// mode 0: accumulate per-channel sum/sumsq of h (BN stats).
// mode 1: recompute h, apply BN affine + leaky, atomic-max over k into xcat.
// ---------------------------------------------------------------------------
__global__ __launch_bounds__(32)
void edge_gemm_kernel(const float* __restrict__ X, int cin, int coffin, int bstride,
                      const int* __restrict__ idx, const unsigned short* __restrict__ WtT,
                      int O, int mode,
                      float* sums, float* sumsq,
                      const float* __restrict__ scale, const float* __restrict__ shift,
                      float* out, int choffOut) {
  const int Ktot   = 2 * cin;
  const int Kpad   = (Ktot + 31) & ~31;
  const int ktiles = Kpad >> 5;
  const int lane   = threadIdx.x;
  __shared__ unsigned short Al[GT][16 * 512];
  __shared__ unsigned short Bl[512 * 16];

  // Build GT A tiles (16 edge rows x Kpad channels each), zero-padded K.
  for (int g = 0; g < GT; ++g) {
    int tile = blockIdx.x * GT + g;
    for (int e = lane; e < 16 * Kpad; e += 32) {
      int r = e / Kpad, c = e - r * Kpad;
      int gr  = tile * 16 + r;
      int bb  = gr / (NPTS * KNN);
      int rem = gr - bb * (NPTS * KNN);
      int n   = rem / KNN;
      int kk  = rem - n * KNN;
      float v = 0.f;
      if (c < Ktot) {
        if (c < cin) {
          int m = idx[((size_t)bb * NPTS + n) * KNN + kk];
          const float* Xc = X + (size_t)bb * bstride + (size_t)(coffin + c) * NPTS;
          v = Xc[m] - Xc[n];                       // neighbor - center
        } else {
          const float* Xc = X + (size_t)bb * bstride + (size_t)(coffin + c - cin) * NPTS;
          v = Xc[n];                               // center
        }
      }
      Al[g][r * Kpad + c] = f2bf(v);
    }
  }
  // Zero-pad B rows in [Ktot, Kpad) once; async copies never touch them.
  for (int e = lane; e < (Kpad - Ktot) * 16; e += 32) Bl[Ktot * 16 + e] = 0;
  __syncthreads();

  const unsigned ldsB = (unsigned)(size_t)&Bl[0];
  const int otiles = O >> 4;
  for (int ot = 0; ot < otiles; ++ot) {
    async_copy_B(WtT, O, ot, ldsB, Ktot, lane);
    __syncthreads();

    int col = lane & 15;
    int o   = ot * 16 + col;
    float s = 0.f, s2 = 0.f;
    float sc = 0.f, sh = 0.f;
    if (mode != 0) { sc = scale[o]; sh = shift[o]; }

    for (int g = 0; g < GT; ++g) {
      v8f acc = {0.f, 0.f, 0.f, 0.f, 0.f, 0.f, 0.f, 0.f};
      for (int kt = 0; kt < ktiles; ++kt)
        acc = wmma_tile(&Al[g][kt * 32], Kpad, Bl + kt * 512, acc, lane);

      if (mode == 0) {
#pragma unroll
        for (int i = 0; i < 8; ++i) { float v = acc[i]; s += v; s2 += v * v; }
      } else {
        int tile = blockIdx.x * GT + g;
#pragma unroll
        for (int i = 0; i < 8; ++i) {
          int Mi  = (lane < 16) ? i : (8 + i);
          int gr  = tile * 16 + Mi;
          int bb  = gr / (NPTS * KNN);
          int rem = gr - bb * (NPTS * KNN);
          int n   = rem / KNN;
          float v = acc[i] * sc + sh;
          v = (v > 0.f) ? v : NEG_SLOPE * v;
          atomicMaxF(&out[(size_t)bb * (CATC * NPTS) + (size_t)(choffOut + o) * NPTS + n], v);
        }
      }
    }
    if (mode == 0) { atomicAdd(&sums[o], s); atomicAdd(&sumsq[o], s2); }
    __syncthreads();
  }
}

// ---------------------------------------------------------------------------
// conv5 (1x1 conv 512 -> 1024 over B*N points), same structure.
// mode 0: BN stats.  mode 1: recompute + affine + leaky, max/sum pool over n.
// ---------------------------------------------------------------------------
__global__ __launch_bounds__(32)
void conv5_gemm_kernel(const float* __restrict__ X, const unsigned short* __restrict__ WtT,
                       int mode, float* sums, float* sumsq,
                       const float* __restrict__ scale, const float* __restrict__ shift,
                       float* fmaxb, float* fsumb) {
  const int Ktot = CATC, Kpad = CATC, ktiles = CATC / 32, O = 1024;
  const int lane = threadIdx.x;
  __shared__ unsigned short Al[GT][16 * 512];
  __shared__ unsigned short Bl[512 * 16];

  for (int g = 0; g < GT; ++g) {
    int tile = blockIdx.x * GT + g;
    for (int e = lane; e < 16 * Kpad; e += 32) {
      int r  = e / Kpad, c = e - r * Kpad;
      int gr = tile * 16 + r;
      int bb = gr / NPTS, n = gr - bb * NPTS;
      Al[g][r * Kpad + c] = f2bf(X[(size_t)bb * (CATC * NPTS) + (size_t)c * NPTS + n]);
    }
  }
  __syncthreads();

  const unsigned ldsB = (unsigned)(size_t)&Bl[0];
  for (int ot = 0; ot < O / 16; ++ot) {
    async_copy_B(WtT, O, ot, ldsB, Ktot, lane);
    __syncthreads();

    int col = lane & 15;
    int o   = ot * 16 + col;
    float s = 0.f, s2 = 0.f;
    float sc = 0.f, sh = 0.f;
    if (mode != 0) { sc = scale[o]; sh = shift[o]; }

    for (int g = 0; g < GT; ++g) {
      v8f acc = {0.f, 0.f, 0.f, 0.f, 0.f, 0.f, 0.f, 0.f};
      for (int kt = 0; kt < ktiles; ++kt)
        acc = wmma_tile(&Al[g][kt * 32], Kpad, Bl + kt * 512, acc, lane);

      if (mode == 0) {
#pragma unroll
        for (int i = 0; i < 8; ++i) { float v = acc[i]; s += v; s2 += v * v; }
      } else {
        int tile = blockIdx.x * GT + g;
#pragma unroll
        for (int i = 0; i < 8; ++i) {
          int Mi = (lane < 16) ? i : (8 + i);
          int gr = tile * 16 + Mi;
          int bb = gr / NPTS;
          float v = acc[i] * sc + sh;
          v = (v > 0.f) ? v : NEG_SLOPE * v;
          atomicMaxF(&fmaxb[bb * 1024 + o], v);
          atomicAdd(&fsumb[bb * 1024 + o], v);
        }
      }
    }
    if (mode == 0) { atomicAdd(&sums[o], s); atomicAdd(&sumsq[o], s2); }
    __syncthreads();
  }
}

__global__ void bn_finalize_kernel(const float* __restrict__ sums, const float* __restrict__ sumsq,
                                   const float* __restrict__ g, const float* __restrict__ be,
                                   float cnt, int O, float* __restrict__ scale,
                                   float* __restrict__ shift) {
  int o = blockIdx.x * blockDim.x + threadIdx.x;
  if (o >= O) return;
  float mean = sums[o] / cnt;
  float var  = sumsq[o] / cnt - mean * mean;
  float sc   = g[o] * rsqrtf(var + BN_EPS);
  scale[o] = sc;
  shift[o] = be[o] - mean * sc;
}

__global__ void build_f_kernel(const float* __restrict__ fmaxb, const float* __restrict__ fsumb,
                               float* __restrict__ f) {
  int t = blockIdx.x * blockDim.x + threadIdx.x;
  if (t >= NB * 1024) return;
  int b = t / 1024, o = t - b * 1024;
  f[b * 2048 + o]        = fmaxb[t];
  f[b * 2048 + 1024 + o] = fsumb[t] * (1.f / NPTS);
}

__global__ void fc_kernel(const float* __restrict__ in, const float* __restrict__ W,
                          const float* __restrict__ bias, float* __restrict__ out,
                          int Cin, int Cout) {
  int t = blockIdx.x * blockDim.x + threadIdx.x;
  if (t >= NB * Cout) return;
  int b = t / Cout, o = t - b * Cout;
  const float* ip = in + (size_t)b * Cin;
  const float* wp = W + (size_t)o * Cin;
  float s = bias ? bias[o] : 0.f;
  for (int c = 0; c < Cin; ++c) s += ip[c] * wp[c];
  out[t] = s;
}

__global__ void bn_batch_kernel(const float* __restrict__ z, const float* __restrict__ g,
                                const float* __restrict__ be, float* __restrict__ out,
                                int C, int doLeaky) {
  int o = blockIdx.x * blockDim.x + threadIdx.x;
  if (o >= C) return;
  float s = 0.f, s2 = 0.f;
  for (int b = 0; b < NB; ++b) { float v = z[b * C + o]; s += v; s2 += v * v; }
  float mean = s * (1.f / NB);
  float var  = s2 * (1.f / NB) - mean * mean;
  float sc   = g[o] * rsqrtf(var + BN_EPS);
  float sh   = be[o] - mean * sc;
  for (int b = 0; b < NB; ++b) {
    float v = z[b * C + o] * sc + sh;
    if (doLeaky) v = (v > 0.f) ? v : NEG_SLOPE * v;
    out[b * C + o] = v;
  }
}

__global__ void l2norm_kernel(const float* __restrict__ f, float* __restrict__ out) {
  __shared__ float red[256];
  int b = blockIdx.x, t = threadIdx.x;
  float s = 0.f;
  for (int c = t; c < 512; c += 256) { float v = f[b * 512 + c]; s += v * v; }
  red[t] = s;
  __syncthreads();
  for (int st = 128; st > 0; st >>= 1) {
    if (t < st) red[t] += red[t + st];
    __syncthreads();
  }
  float nrm = fmaxf(sqrtf(red[0]), 1e-12f);
  for (int c = t; c < 512; c += 256) out[b * 512 + c] = f[b * 512 + c] / nrm;
}

// ---------------------------------------------------------------------------
extern "C" void kernel_launch(void* const* d_in, const int* in_sizes, int n_in,
                              void* d_out, int out_size, void* d_ws, size_t ws_size,
                              hipStream_t stream) {
  (void)in_sizes; (void)n_in; (void)out_size; (void)ws_size;
  const float* x   = (const float*)d_in[0];
  const float* W1  = (const float*)d_in[1];
  const float* g1  = (const float*)d_in[2];
  const float* be1 = (const float*)d_in[3];
  const float* W2  = (const float*)d_in[4];
  const float* g2  = (const float*)d_in[5];
  const float* be2 = (const float*)d_in[6];
  const float* W3  = (const float*)d_in[7];
  const float* g3  = (const float*)d_in[8];
  const float* be3 = (const float*)d_in[9];
  const float* W4  = (const float*)d_in[10];
  const float* g4  = (const float*)d_in[11];
  const float* be4 = (const float*)d_in[12];
  const float* W5  = (const float*)d_in[13];
  const float* g5  = (const float*)d_in[14];
  const float* be5 = (const float*)d_in[15];
  const float* Wl1 = (const float*)d_in[16];
  const float* gb1 = (const float*)d_in[17];
  const float* bb1 = (const float*)d_in[18];
  const float* Wl2 = (const float*)d_in[19];
  const float* bl2 = (const float*)d_in[20];
  const float* gb2 = (const float*)d_in[21];
  const float* bb2 = (const float*)d_in[22];
  const float* Wl3 = (const float*)d_in[23];
  const float* bl3 = (const float*)d_in[24];
  const float* gb3 = (const float*)d_in[25];
  const float* bb3 = (const float*)d_in[26];

  // Workspace carve-up (~36 MB; fits in MI455X's 192 MB L2)
  float* wsf   = (float*)d_ws;
  float* x0    = wsf;                 // 8*3*2048
  float* xcat  = x0 + NB * 3 * NPTS;  // 8*512*2048
  float* sums  = xcat + (size_t)NB * CATC * NPTS;
  float* sumsq = sums + 1024;
  float* scale = sumsq + 1024;
  float* shift = scale + 1024;
  float* fmaxb = shift + 1024;        // 8*1024
  float* fsumb = fmaxb + NB * 1024;   // 8*1024
  float* fvec  = fsumb + NB * 1024;   // 8*2048
  float* fcA   = fvec + NB * 2048;
  float* fcB   = fcA + 16384;
  float* fcC   = fcB + 16384;
  int*   idxb  = (int*)(fcC + 16384);                 // 8*2048*20
  unsigned short* wbf = (unsigned short*)(idxb + NB * NPTS * KNN); // up to 512*1024

  const int threads = 256;
  const int edgeTiles  = (NB * NPTS * KNN) / 16;  // 20480
  const int conv5Tiles = (NB * NPTS) / 16;        // 1024

  xpose_kernel<<<(NB * NPTS + threads - 1) / threads, threads, 0, stream>>>(x, x0);

  struct Layer { const float* X; int cin, coffin, bstride;
                 const float* W; const float* g; const float* be; int O, choff; };
  Layer L[4] = {
      {x0,   3,   0,   3 * NPTS,    W1, g1, be1, 64,  0},
      {xcat, 64,  0,   CATC * NPTS, W2, g2, be2, 64,  64},
      {xcat, 64,  64,  CATC * NPTS, W3, g3, be3, 128, 128},
      {xcat, 128, 128, CATC * NPTS, W4, g4, be4, 256, 256},
  };

  for (int l = 0; l < 4; ++l) {
    const Layer& P = L[l];
    knn_kernel<<<NB * NPTS, 32, 0, stream>>>(P.X, P.cin, P.coffin, P.bstride, idxb);
    int wn = P.O * 2 * P.cin;
    cvtw_t_kernel<<<(wn + threads - 1) / threads, threads, 0, stream>>>(P.W, wbf, P.O, 2 * P.cin);
    fill_kernel<<<(2048 + threads - 1) / threads, threads, 0, stream>>>(sums, 0.f, 2048);
    for (int b = 0; b < NB; ++b)
      fill_kernel<<<(P.O * NPTS + threads - 1) / threads, threads, 0, stream>>>(
          xcat + (size_t)b * CATC * NPTS + (size_t)P.choff * NPTS, NEG_INF, P.O * NPTS);
    // pass 1: BN stats via WMMA
    edge_gemm_kernel<<<edgeTiles / GT, 32, 0, stream>>>(P.X, P.cin, P.coffin, P.bstride, idxb,
                                                        wbf, P.O, 0, sums, sumsq,
                                                        nullptr, nullptr, nullptr, 0);
    bn_finalize_kernel<<<(P.O + 63) / 64, 64, 0, stream>>>(
        sums, sumsq, P.g, P.be, (float)(NB * NPTS * KNN), P.O, scale, shift);
    // pass 2: recompute + affine + leaky + max over k
    edge_gemm_kernel<<<edgeTiles / GT, 32, 0, stream>>>(P.X, P.cin, P.coffin, P.bstride, idxb,
                                                        wbf, P.O, 1, nullptr, nullptr,
                                                        scale, shift, xcat, P.choff);
  }

  // conv5: 512 -> 1024
  cvtw_t_kernel<<<(1024 * 512 + threads - 1) / threads, threads, 0, stream>>>(W5, wbf, 1024, 512);
  fill_kernel<<<(2048 + threads - 1) / threads, threads, 0, stream>>>(sums, 0.f, 2048);
  fill_kernel<<<(NB * 1024 + threads - 1) / threads, threads, 0, stream>>>(fmaxb, NEG_INF, NB * 1024);
  fill_kernel<<<(NB * 1024 + threads - 1) / threads, threads, 0, stream>>>(fsumb, 0.f, NB * 1024);
  conv5_gemm_kernel<<<conv5Tiles / GT, 32, 0, stream>>>(xcat, wbf, 0, sums, sumsq,
                                                        nullptr, nullptr, nullptr, nullptr);
  bn_finalize_kernel<<<(1024 + 63) / 64, 64, 0, stream>>>(
      sums, sumsq, g5, be5, (float)(NB * NPTS), 1024, scale, shift);
  conv5_gemm_kernel<<<conv5Tiles / GT, 32, 0, stream>>>(xcat, wbf, 1, nullptr, nullptr,
                                                        scale, shift, fmaxb, fsumb);
  build_f_kernel<<<(NB * 1024 + threads - 1) / threads, threads, 0, stream>>>(fmaxb, fsumb, fvec);

  // FC head (tiny: B=8)
  fc_kernel<<<(NB * 512 + threads - 1) / threads, threads, 0, stream>>>(fvec, Wl1, nullptr, fcA, 2048, 512);
  bn_batch_kernel<<<(512 + 63) / 64, 64, 0, stream>>>(fcA, gb1, bb1, fcB, 512, 1);
  fc_kernel<<<(NB * 256 + threads - 1) / threads, threads, 0, stream>>>(fcB, Wl2, bl2, fcC, 512, 256);
  bn_batch_kernel<<<(256 + 63) / 64, 64, 0, stream>>>(fcC, gb2, bb2, fcA, 256, 1);
  fc_kernel<<<(NB * 512 + threads - 1) / threads, threads, 0, stream>>>(fcA, Wl3, bl3, fcB, 256, 512);
  bn_batch_kernel<<<(512 + 63) / 64, 64, 0, stream>>>(fcB, gb3, bb3, fcC, 512, 0);
  l2norm_kernel<<<NB, 256, 0, stream>>>(fcC, (float*)d_out);
}
